// SelfAttention_28157805593230
// MI455X (gfx1250) — matrix operand order repaired and checked by reading, not compiled, not run on hardware
//
#include <hip/hip_runtime.h>

typedef __attribute__((ext_vector_type(16))) _Float16 v16h;
typedef __attribute__((ext_vector_type(8)))  _Float16 v8h;
typedef __attribute__((ext_vector_type(8)))  float    v8f;

static constexpr int Bn = 4;
static constexpr int Cn = 256;
static constexpr int Dn = 32;     // C/8
static constexpr int Ln = 4096;

// ---------------- f32 -> f16 convert (weights) ----------------
__global__ void cvt_f32_f16(const float* __restrict__ src, _Float16* __restrict__ dst, int n) {
  int i = blockIdx.x * blockDim.x + threadIdx.x;
  if (i < n) dst[i] = (_Float16)src[i];
}

// ---------------- x [B][C][L] f32 -> xt [B][L][C] f16 (tiled transpose) ----------------
__global__ __launch_bounds__(256)
void cvt_transpose(const float* __restrict__ src, _Float16* __restrict__ dst) {
  __shared__ _Float16 tile[32][33];
  const int b  = blockIdx.z;
  const int l0 = blockIdx.x * 32;
  const int c0 = blockIdx.y * 32;
  const int tx = threadIdx.x & 31;
  const int ty = threadIdx.x >> 5;   // 0..7
#pragma unroll
  for (int i = 0; i < 32; i += 8)
    tile[ty + i][tx] = (_Float16)src[(size_t)b * Cn * Ln + (size_t)(c0 + ty + i) * Ln + l0 + tx];
  __syncthreads();
#pragma unroll
  for (int i = 0; i < 32; i += 8)
    dst[(size_t)b * Ln * Cn + (size_t)(l0 + ty + i) * Cn + c0 + tx] = tile[tx][ty + i];
}

// ---------------- projection GEMM: out[o,l] = sum_c w[o,c]*xt[l,c] + bias[o] ----------------
// one wave computes a 16(o) x 16(l) tile; all WMMA operands are contiguous vector loads.
// tpose==0: write out[b][o][L] (V). tpose==1: write out[b][l][O] (Q,K).
__global__ __launch_bounds__(32)
void proj_wmma(const _Float16* __restrict__ wh, const float* __restrict__ bias,
               const _Float16* __restrict__ xt, _Float16* __restrict__ outh,
               int O, int tpose) {
  const int lane = threadIdx.x & 31;
  const int n16  = lane & 15;
  const bool lo  = (lane < 16);
  const int hi8  = lo ? 0 : 8;
  const int lo8  = lo ? 0 : 8;    // unused; A and B both use lo16
  const int lo16 = lo ? 0 : 16;   // B-layout K base

  const int l0 = blockIdx.x * 16;
  const int o0 = blockIdx.y * 16;
  const int b  = blockIdx.z;

  const _Float16* xb = xt + (size_t)b * Ln * Cn + (size_t)(l0 + n16) * Cn;
  const _Float16* wr = wh + (size_t)(o0 + n16) * Cn;

  v8f acc = {};
  for (int c = 0; c < Cn; c += 32) {
    // A (16x32 f16): one contiguous 16-half run per lane (k = 16h+s)
    v8h a0 = *(const v8h*)&wr[c + lo16];
    v8h a1 = *(const v8h*)&wr[c + lo16 + 8];
    v16h a = __builtin_shufflevector(a0, a1, 0,1,2,3,4,5,6,7,8,9,10,11,12,13,14,15);
    // B (32x16 f16): one contiguous 16-half run per lane
    v16h bm = *(const v16h*)&xb[c + lo16];
    acc = __builtin_amdgcn_wmma_f32_16x16x32_f16(false, a, false, bm, (short)0, acc, false, false);
  }
  if (tpose) {
    v8h res;
#pragma unroll
    for (int r = 0; r < 8; ++r) res[r] = (_Float16)(acc[r] + bias[o0 + r + hi8]);
    *(v8h*)&outh[(size_t)b * Ln * O + (size_t)(l0 + n16) * O + o0 + hi8] = res;
  } else {
    _Float16* ob = outh + (size_t)b * O * Ln;
#pragma unroll
    for (int r = 0; r < 8; ++r)
      ob[(size_t)(o0 + r + hi8) * Ln + l0 + n16] = (_Float16)(acc[r] + bias[o0 + r + hi8]);
  }
}

// ---------------- fused flash attention + residual ----------------
// Qt/Kt are [b][l][32] f16; Vh is [b][c][L] f16.
__global__ __launch_bounds__(128)
void flash_attn(const _Float16* __restrict__ Qt, const _Float16* __restrict__ Kt,
                const _Float16* __restrict__ Vh, const float* __restrict__ x,
                const float* __restrict__ gamma, float* __restrict__ out) {
  __shared__ __align__(32) _Float16 ldsK[32 * 32];    // [j][d]
  __shared__ __align__(32) _Float16 ldsV[256 * 32];   // [c][j]
  __shared__ __align__(32) _Float16 ldsP[4][16 * 32]; // per-wave P tile [i][j]

  const int tid  = threadIdx.x;
  const int w    = tid >> 5;
  const int lane = tid & 31;
  const int n16  = lane & 15;
  const bool lo  = (lane < 16);
  const int hi8  = lo ? 0 : 8;
  const int lo8  = lo ? 0 : 8;
  const int lo16 = lo ? 0 : 16;

  const int b  = blockIdx.y;
  const int i0 = (blockIdx.x * 4 + w) * 16;
  const float scale = 0.17677669529663687f;   // 1/sqrt(32)

  // Q A-tile (16 i x 32 d) from Qt[b][i][d]: two 16B loads
  {
  }
  const _Float16* qrow = Qt + (size_t)b * Ln * Dn + (size_t)(i0 + n16) * Dn;
  v8h q0 = *(const v8h*)&qrow[lo16];
  v8h q1 = *(const v8h*)&qrow[lo16 + 8];
  v16h qa = __builtin_shufflevector(q0, q1, 0,1,2,3,4,5,6,7,8,9,10,11,12,13,14,15);

  v8f o[16] = {};
  float mrow[8], lrow[8];
#pragma unroll
  for (int r = 0; r < 8; ++r) { mrow[r] = -1e30f; lrow[r] = 0.0f; }

  // per-thread async staging addresses
  const int vc   = tid >> 2;        // base c row handled by this thread (+32 per step)
  const int vpart = tid & 3;        // 8-half part within a 32-half row

  for (int jb = 0; jb < Ln; jb += 32) {
    __syncthreads();   // all waves done reading LDS from previous iteration

    // --- async stage K chunk: linear 2KB copy Kt[b][jb..jb+32][0..32] -> ldsK[j][d]
    {
      const _Float16* g = Kt + (size_t)b * Ln * Dn + (size_t)jb * Dn + tid * 8;
      unsigned loff = (unsigned)(size_t)&ldsK[tid * 8];
      asm volatile("global_load_async_to_lds_b128 %0, %1, off"
                   :: "v"(loff), "v"(g) : "memory");
    }
    // --- async stage V chunk: Vh[b][c][jb..jb+32] -> ldsV[c][j], 16KB
#pragma unroll
    for (int k = 0; k < 8; ++k) {
      int c = vc + k * 32;
      const _Float16* g = Vh + (size_t)b * Cn * Ln + (size_t)c * Ln + jb + vpart * 8;
      unsigned loff = (unsigned)(size_t)&ldsV[c * 32 + vpart * 8];
      asm volatile("global_load_async_to_lds_b128 %0, %1, off"
                   :: "v"(loff), "v"(g) : "memory");
    }
    asm volatile("s_wait_asynccnt 0x0" ::: "memory");
    __syncthreads();

    // S = Qt*K : B operands are contiguous 32B LDS loads
    v16h kb0 = *(const v16h*)&ldsK[n16 * 32 + lo16];
    v16h kb1 = *(const v16h*)&ldsK[(16 + n16) * 32 + lo16];
    v8f s0 = {}, s1 = {};
    s0 = __builtin_amdgcn_wmma_f32_16x16x32_f16(false, qa, false, kb0, (short)0, s0, false, false);
    s1 = __builtin_amdgcn_wmma_f32_16x16x32_f16(false, qa, false, kb1, (short)0, s1, false, false);

    // online softmax per row (rows live across 16-lane halves)
    float alpha[8];
#pragma unroll
    for (int r = 0; r < 8; ++r) {
      float a0 = s0[r] * scale, a1 = s1[r] * scale;
      float v = fmaxf(a0, a1);
      v = fmaxf(v, __shfl_xor(v, 1, 32));
      v = fmaxf(v, __shfl_xor(v, 2, 32));
      v = fmaxf(v, __shfl_xor(v, 4, 32));
      v = fmaxf(v, __shfl_xor(v, 8, 32));
      float mn = fmaxf(mrow[r], v);
      float al = __expf(mrow[r] - mn);
      float p0 = __expf(a0 - mn);
      float p1 = __expf(a1 - mn);
      float rs = p0 + p1;
      rs += __shfl_xor(rs, 1, 32);
      rs += __shfl_xor(rs, 2, 32);
      rs += __shfl_xor(rs, 4, 32);
      rs += __shfl_xor(rs, 8, 32);
      lrow[r] = lrow[r] * al + rs;
      mrow[r] = mn;
      alpha[r] = al;
      ldsP[w][(r + hi8) * 32 + n16]      = (_Float16)p0;   // C/D layout -> [i][j]
      ldsP[w][(r + hi8) * 32 + 16 + n16] = (_Float16)p1;
    }
#pragma unroll
    for (int t = 0; t < 16; ++t)
#pragma unroll
      for (int r = 0; r < 8; ++r) o[t][r] *= alpha[r];

    __syncthreads();   // make P visible across lanes (and keep waves in lockstep)

    // O += P * V^T : pa two 16B LDS loads, each vb one 32B LDS load
    v8h p0v = *(const v8h*)&ldsP[w][n16 * 32 + lo16];
    v8h p1v = *(const v8h*)&ldsP[w][n16 * 32 + lo16 + 8];
    v16h pa = __builtin_shufflevector(p0v, p1v, 0,1,2,3,4,5,6,7,8,9,10,11,12,13,14,15);
#pragma unroll
    for (int t = 0; t < 16; ++t) {
      v16h vb = *(const v16h*)&ldsV[(t * 16 + n16) * 32 + lo16];
      o[t] = __builtin_amdgcn_wmma_f32_16x16x32_f16(false, pa, false, vb, (short)0, o[t], false, false);
    }
  }

  // epilogue: out = gamma * (O / l) + x
  const float g = gamma[0];
  const float* xb = x + (size_t)b * Cn * Ln;
  float* ob = out + (size_t)b * Cn * Ln;
#pragma unroll
  for (int r = 0; r < 8; ++r) {
    float inv = 1.0f / lrow[r];
    int i = i0 + r + hi8;
#pragma unroll
    for (int t = 0; t < 16; ++t) {
      int c = t * 16 + n16;
      ob[(size_t)c * Ln + i] = g * o[t][r] * inv + xb[(size_t)c * Ln + i];
    }
  }
}

extern "C" void kernel_launch(void* const* d_in, const int* in_sizes, int n_in,
                              void* d_out, int out_size, void* d_ws, size_t ws_size,
                              hipStream_t stream) {
  (void)in_sizes; (void)n_in; (void)out_size; (void)ws_size;
  const float* x     = (const float*)d_in[0];
  const float* wq    = (const float*)d_in[1];
  const float* bq    = (const float*)d_in[2];
  const float* wk    = (const float*)d_in[3];
  const float* bk    = (const float*)d_in[4];
  const float* wv    = (const float*)d_in[5];
  const float* bv    = (const float*)d_in[6];
  const float* gamma = (const float*)d_in[7];
  float* out = (float*)d_out;

  char* ws = (char*)d_ws;
  _Float16* xt  = (_Float16*)(ws);                                   // 8 MB  [b][l][c]
  _Float16* Vh  = (_Float16*)(ws + (size_t)(8u << 20));              // 8 MB  [b][c][L]
  _Float16* Qt  = (_Float16*)(ws + (size_t)(16u << 20));             // 1 MB  [b][l][32]
  _Float16* Kt  = (_Float16*)(ws + (size_t)(17u << 20));             // 1 MB  [b][l][32]
  _Float16* wqh = (_Float16*)(ws + (size_t)(18u << 20));             // 16 KB
  _Float16* wkh = (_Float16*)(ws + (size_t)(18u << 20) + 32 * 1024);
  _Float16* wvh = (_Float16*)(ws + (size_t)(18u << 20) + 64 * 1024); // 128 KB

  dim3 gt(Ln / 32, Cn / 32, Bn);
  cvt_transpose<<<gt, 256, 0, stream>>>(x, xt);
  cvt_f32_f16<<<(Dn * Cn + 255) / 256, 256, 0, stream>>>(wq, wqh, Dn * Cn);
  cvt_f32_f16<<<(Dn * Cn + 255) / 256, 256, 0, stream>>>(wk, wkh, Dn * Cn);
  cvt_f32_f16<<<(Cn * Cn + 255) / 256, 256, 0, stream>>>(wv, wvh, Cn * Cn);

  dim3 gqk(Ln / 16, Dn / 16, Bn);
  proj_wmma<<<gqk, 32, 0, stream>>>(wqh, bq, xt, Qt, Dn, 1);   // Q transposed [l][32]
  proj_wmma<<<gqk, 32, 0, stream>>>(wkh, bk, xt, Kt, Dn, 1);   // K transposed [l][32]
  dim3 gv(Ln / 16, Cn / 16, Bn);
  proj_wmma<<<gv, 32, 0, stream>>>(wvh, bv, xt, Vh, Cn, 0);    // V normal [c][L]

  dim3 ga(Ln / 64, Bn);
  flash_attn<<<ga, 128, 0, stream>>>(Qt, Kt, Vh, x, gamma, out);
}
